// GAttnEncoderLayer_26920855011381
// MI455X (gfx1250) — matrix operand order repaired and checked by reading, not compiled, not run on hardware
//
#include <hip/hip_runtime.h>

#define BB   4
#define NN   2048
#define DD   256
#define KKC  16
#define NPAD 2064          // 8 left pad + 2048 + 8 right pad
#define DKK  (DD * KKC)    // 4096
#define EPSLN 1e-5f
#define LEAK 0.01f

typedef __attribute__((ext_vector_type(16))) __bf16 v16bf;
typedef __attribute__((ext_vector_type(8)))  __bf16 v8bf;
typedef __attribute__((ext_vector_type(8)))  float  v8f;

#if __has_builtin(__builtin_amdgcn_tensor_load_to_lds) && \
    __has_builtin(__builtin_amdgcn_s_wait_tensorcnt)
#define GATTN_HAS_TDM 1
typedef __attribute__((ext_vector_type(4))) unsigned int tdm_v4u;
typedef __attribute__((ext_vector_type(8))) int          tdm_v8i;
typedef __attribute__((ext_vector_type(4))) int          tdm_v4i;
#endif

// ---------------------------------------------------------------------------
// WMMA fragment helpers (wave32; A/B k order is this code's contiguous choice, C/D per CDNA5 ISA)
// A (16x32 bf16): lane l holds row M=l%16; elem j -> K = 16*(l/16)+j
//   -> two contiguous 8-elem (16B) runs: vectorizes to 2 x b128.
// B (32x16 bf16): lane l holds col N=l%16; elem j -> K = 16*(l/16)+j.
//   Operand stored K-major (Bt[col][K]) -> 16 contiguous elems = 1 x v16bf.
// C/D (16x16 f32): lane l holds col N=l%16; elem r -> M = r + 8*(l/16)
// ---------------------------------------------------------------------------
__device__ __forceinline__ v16bf load_a_frag(const __bf16* __restrict__ A, int lda,
                                             int row0, int col0, int lane) {
  const __bf16* p = A + (size_t)(row0 + (lane & 15)) * lda + col0 + ((lane >> 4) << 4);
  v8bf lo = *(const v8bf*)p;
  v8bf hi = *(const v8bf*)(p + 8);
  return __builtin_shufflevector(lo, hi, 0, 1, 2, 3, 4, 5, 6, 7,
                                 8, 9, 10, 11, 12, 13, 14, 15);
}

// Bt is the K-major (transposed) view: element j = Bt[col0 + lane%16][k0 + 16*half + j]
__device__ __forceinline__ v16bf load_bT_frag(const __bf16* __restrict__ Bt, int ldb,
                                              int k0, int col0, int lane) {
  const __bf16* p = Bt + (size_t)(col0 + (lane & 15)) * ldb + k0 + ((lane >> 4) << 4);
  return *(const v16bf*)p;
}

__device__ __forceinline__ v8f wmma_bf16(v16bf a, v16bf b, v8f c) {
  return __builtin_amdgcn_wmma_f32_16x16x32_bf16(false, a, false, b,
                                                 (short)0, c, false, false);
}

__device__ __forceinline__ v8f zero8() {
  v8f z = {0.f, 0.f, 0.f, 0.f, 0.f, 0.f, 0.f, 0.f};
  return z;
}

// ---------------------------------------------------------------------------
// Packing kernels
// ---------------------------------------------------------------------------
// src[B,N,D] -> xpadT[B,NPAD,D] bf16 (row n+8, feature-contiguous, zero pad)
__global__ void gattn_pack_x(const float* __restrict__ src, __bf16* __restrict__ xpadT) {
  int idx = blockIdx.x * blockDim.x + threadIdx.x;
  int b = idx / (NPAD * DD);
  int rem = idx - b * (NPAD * DD);
  int p = rem / DD;
  int c = rem - p * DD;
  int n = p - 8;
  float v = 0.f;
  if (n >= 0 && n < NN) v = src[((size_t)b * NN + n) * DD + c];
  xpadT[idx] = (__bf16)v;
}

// W[o,i,kc] (D,D,K) -> Wpack[o, kc*D + i]  (A operand, row-major)
__global__ void gattn_pack_convw(const float* __restrict__ W, __bf16* __restrict__ Wp) {
  int idx = blockIdx.x * blockDim.x + threadIdx.x;
  int o = idx / DKK;
  int rem = idx - o * DKK;
  int kc = rem / DD;
  int i = rem - kc * DD;
  Wp[idx] = (__bf16)W[((size_t)o * DD + i) * KKC + kc];
}

// plain fp32 -> bf16 copy (MLP weights keep [dout,din] = K-major B layout)
__global__ void gattn_pack_f2b(const float* __restrict__ W, __bf16* __restrict__ Wb) {
  int idx = blockIdx.x * blockDim.x + threadIdx.x;
  Wb[idx] = (__bf16)W[idx];
}

// ---------------------------------------------------------------------------
// Conv1d QKV as GEMM: q[o,n] = sum_{kc,i} Wq[o,i,kc] * x[i, n+kc-7]
// Block = 8 waves sharing (b, o-tile); weights staged in LDS per tap.
// Outputs: qT[B,N,D], kT[B,N,D], vmat[B,D,N] (bf16)
// ---------------------------------------------------------------------------
__global__ __launch_bounds__(256) void gattn_conv_qkv(
    const __bf16* __restrict__ xpadT,
    const __bf16* __restrict__ wpq, const __bf16* __restrict__ wpk,
    const __bf16* __restrict__ wpv,
    const float* __restrict__ bq, const float* __restrict__ bk,
    const float* __restrict__ bv,
    __bf16* __restrict__ qT, __bf16* __restrict__ kT, __bf16* __restrict__ vmat) {
  int wid = blockIdx.x * 8 + (threadIdx.x >> 5);
  int lane = threadIdx.x & 31;
  int b = wid >> 11;                 // 16 o-tiles * 128 n-tiles per batch
  int rem = wid & 2047;
  int o0 = (rem >> 7) << 4;          // uniform across the block
  int n0 = (rem & 127) << 4;

  __shared__ __bf16 wqs[16 * DD], wks[16 * DD], wvs[16 * DD];  // 24 KB

  const __bf16* xp = xpadT + (size_t)b * NPAD * DD;
  v8f accq = zero8(), acck = zero8(), accv = zero8();

  for (int kc = 0; kc < KKC; ++kc) {
    __syncthreads();
    // stage 16x256 weight tiles for this tap (coalesced uint4)
    {
      int i0 = threadIdx.x * 2;
#pragma unroll
      for (int t = 0; t < 2; ++t) {
        int idx = i0 + t;                 // 0..511 uint4 slots
        int row = idx >> 5;               // 32 uint4 per 256-elem row
        int cq = (idx & 31) << 3;         // elem offset in row
        size_t g = (size_t)(o0 + row) * DKK + kc * DD + cq;
        *(uint4*)&wqs[(size_t)row * DD + cq] = *(const uint4*)&wpq[g];
        *(uint4*)&wks[(size_t)row * DD + cq] = *(const uint4*)&wpk[g];
        *(uint4*)&wvs[(size_t)row * DD + cq] = *(const uint4*)&wpv[g];
      }
    }
    __syncthreads();
    if (kc + 1 < KKC)   // warm L2 for next tap's weights (global_prefetch_b8)
      __builtin_prefetch(&wpq[(size_t)(o0 + (threadIdx.x & 15)) * DKK + (kc + 1) * DD], 0, 0);
#pragma unroll
    for (int ic = 0; ic < 8; ++ic) {
      int i0 = ic * 32;
      v16bf bf = load_bT_frag(xp, DD, i0, n0 + kc + 1, lane);
      v16bf aq = load_a_frag(wqs, DD, 0, i0, lane);
      v16bf ak = load_a_frag(wks, DD, 0, i0, lane);
      v16bf av = load_a_frag(wvs, DD, 0, i0, lane);
      accq = wmma_bf16(aq, bf, accq);
      acck = wmma_bf16(ak, bf, acck);
      accv = wmma_bf16(av, bf, accv);
    }
  }

  int col = lane & 15, half = lane >> 4;
  int n = n0 + col;
  v8bf vq, vk;
#pragma unroll
  for (int r = 0; r < 8; ++r) {
    int o = o0 + r + (half << 3);
    vq[r] = (__bf16)(accq[r] + bq[o]);
    vk[r] = (__bf16)(acck[r] + bk[o]);
    vmat[((size_t)b * DD + o) * NN + n] = (__bf16)(accv[r] + bv[o]);
  }
  *(v8bf*)&qT[((size_t)b * NN + n) * DD + o0 + (half << 3)] = vq;
  *(v8bf*)&kT[((size_t)b * NN + n) * DD + o0 + (half << 3)] = vk;
}

// ---------------------------------------------------------------------------
// Fused scores + mask + softmax + state update.
// Block = 512 threads = 16 waves; q-band staged in LDS once.
// ---------------------------------------------------------------------------
__global__ __launch_bounds__(512) void gattn_scores_softmax(
    const __bf16* __restrict__ qT, const __bf16* __restrict__ kT,
    const float* __restrict__ dis, const int* __restrict__ adj,
    const float* __restrict__ x_lst,
    __bf16* __restrict__ attn, float* __restrict__ xlst_out) {
  int b = blockIdx.x >> 7;
  int n0 = (blockIdx.x & 127) << 4;
  int w = threadIdx.x >> 5;
  int lane = threadIdx.x & 31;
  int col = lane & 15, half = lane >> 4;

  const __bf16* q = qT + (size_t)b * NN * DD;
  const __bf16* k = kT + (size_t)b * NN * DD;

  __shared__ __bf16 qs[16 * DD];   // 8 KB, whole q band is contiguous
  ((uint4*)qs)[threadIdx.x] = ((const uint4*)(q + (size_t)n0 * DD))[threadIdx.x];
  __syncthreads();

  v8f acc[8];
#pragma unroll
  for (int mt = 0; mt < 8; ++mt) acc[mt] = zero8();

  for (int ch = 0; ch < 8; ++ch) {
    v16bf a = load_a_frag(qs, DD, 0, ch * 32, lane);
#pragma unroll
    for (int mt = 0; mt < 8; ++mt) {
      int m0 = ((w << 3) + mt) << 4;
      v16bf bf = load_bT_frag(k, DD, ch * 32, m0, lane);
      acc[mt] = wmma_bf16(a, bf, acc[mt]);
    }
  }

  const float scale = 0.0625f;  // 1/sqrt(256)
#pragma unroll
  for (int mt = 0; mt < 8; ++mt) {
    int m = (((w << 3) + mt) << 4) + col;
#pragma unroll
    for (int r = 0; r < 8; ++r) {
      int nrow = n0 + r + (half << 3);
      size_t idx = ((size_t)b * NN + nrow) * NN + m;
      float s = acc[mt][r] * scale - dis[idx];
      if (adj[idx] <= 0) s = -1e9f;
      acc[mt][r] = s;
    }
  }

  __shared__ float red[16][16];
  __shared__ float rowstat[16];

  // ---- row max ----
  float rmax[8];
#pragma unroll
  for (int r = 0; r < 8; ++r) {
    float m = acc[0][r];
#pragma unroll
    for (int mt = 1; mt < 8; ++mt) m = fmaxf(m, acc[mt][r]);
    for (int off = 1; off < 16; off <<= 1) m = fmaxf(m, __shfl_xor(m, off, 16));
    rmax[r] = m;
  }
  if (col == 0)
#pragma unroll
    for (int r = 0; r < 8; ++r) red[(half << 3) + r][w] = rmax[r];
  __syncthreads();
  if (threadIdx.x < 16) {
    float m = red[threadIdx.x][0];
    for (int i = 1; i < 16; ++i) m = fmaxf(m, red[threadIdx.x][i]);
    rowstat[threadIdx.x] = m;
  }
  __syncthreads();
  float rowmax[8];
#pragma unroll
  for (int r = 0; r < 8; ++r) rowmax[r] = rowstat[(half << 3) + r];
  __syncthreads();

  // ---- row sum of exp ----
  float rsum[8];
#pragma unroll
  for (int r = 0; r < 8; ++r) {
    float s = 0.f;
#pragma unroll
    for (int mt = 0; mt < 8; ++mt) s += __expf(acc[mt][r] - rowmax[r]);
    for (int off = 1; off < 16; off <<= 1) s += __shfl_xor(s, off, 16);
    rsum[r] = s;
  }
  if (col == 0)
#pragma unroll
    for (int r = 0; r < 8; ++r) red[(half << 3) + r][w] = rsum[r];
  __syncthreads();
  if (threadIdx.x < 16) {
    float s = red[threadIdx.x][0];
    for (int i = 1; i < 16; ++i) s += red[threadIdx.x][i];
    rowstat[threadIdx.x] = s;
  }
  __syncthreads();

  // ---- emit probabilities + accumulated attention state ----
#pragma unroll
  for (int r = 0; r < 8; ++r) {
    float inv = 1.0f / rowstat[(half << 3) + r];
    int nrow = n0 + r + (half << 3);
#pragma unroll
    for (int mt = 0; mt < 8; ++mt) {
      int m = (((w << 3) + mt) << 4) + col;
      size_t idx = ((size_t)b * NN + nrow) * NN + m;
      float p = __expf(acc[mt][r] - rowmax[r]) * inv;
      attn[idx] = (__bf16)p;
      xlst_out[idx] = x_lst[idx] + p;
    }
  }
}

// ---------------------------------------------------------------------------
// x1[n,dd] = sum_m attn[n,m] * v[dd,m]; attn band staged in LDS per 256-col slab
// ---------------------------------------------------------------------------
__global__ __launch_bounds__(256) void gattn_attn_v(
    const __bf16* __restrict__ attn, const __bf16* __restrict__ vmat,
    float* __restrict__ x1) {
  int wid = blockIdx.x * 8 + (threadIdx.x >> 5);
  int lane = threadIdx.x & 31;
  int b = wid >> 11;
  int rem = wid & 2047;
  int n0 = (rem >> 4) << 4;          // uniform across block
  int d0 = (rem & 15) << 4;

  const __bf16* a = attn + (size_t)b * NN * NN;
  const __bf16* v = vmat + (size_t)b * DD * NN;

  __shared__ __bf16 as[16 * 256];    // 8 KB slab of the attn band

  v8f acc = zero8();
  for (int mc = 0; mc < 8; ++mc) {
    __syncthreads();
    {
      const __bf16* gsrc = a + (size_t)n0 * NN + mc * 256;
      int i0 = threadIdx.x * 2;
#pragma unroll
      for (int t = 0; t < 2; ++t) {
        int idx = i0 + t;             // 0..511 uint4 slots
        int row = idx >> 5;
        int cq = (idx & 31) << 3;
        *(uint4*)&as[(size_t)row * 256 + cq] = *(const uint4*)(gsrc + (size_t)row * NN + cq);
      }
    }
    __syncthreads();
#pragma unroll
    for (int ch = 0; ch < 8; ++ch) {
      v16bf af = load_a_frag(as, 256, 0, ch * 32, lane);
      v16bf bf = load_bT_frag(v, NN, mc * 256 + ch * 32, d0, lane);
      acc = wmma_bf16(af, bf, acc);
    }
  }
  int col = lane & 15, half = lane >> 4;
#pragma unroll
  for (int r = 0; r < 8; ++r) {
    int n = n0 + r + (half << 3);
    x1[((size_t)b * NN + n) * DD + d0 + col] = acc[r];
  }
}

// ---------------------------------------------------------------------------
// LayerNorm over d=256
// ---------------------------------------------------------------------------
__global__ __launch_bounds__(256) void gattn_layernorm(
    const float* __restrict__ xa, const float* __restrict__ xb,
    const float* __restrict__ g, const float* __restrict__ be,
    float* __restrict__ outf, __bf16* __restrict__ outb) {
  int row = blockIdx.x;
  int c = threadIdx.x;
  size_t base = (size_t)row * DD;
  float v = xa[base + c] + xb[base + c];

  __shared__ float sb[DD];
  sb[c] = v;
  __syncthreads();
  for (int s = DD / 2; s > 0; s >>= 1) {
    if (c < s) sb[c] += sb[c + s];
    __syncthreads();
  }
  float mu = sb[0] * (1.0f / DD);
  __syncthreads();
  float d = v - mu;
  sb[c] = d * d;
  __syncthreads();
  for (int s = DD / 2; s > 0; s >>= 1) {
    if (c < s) sb[c] += sb[c + s];
    __syncthreads();
  }
  float var = sb[0] * (1.0f / DD);
  float y = d * rsqrtf(var + EPSLN) * g[c] + be[c];
  if (outf) outf[base + c] = y;
  if (outb) outb[base + c] = (__bf16)y;
}

// ---------------------------------------------------------------------------
// MLP GEMM: out[r,c] = act(A[r,:] @ W[c,:] + bias[c]); A band staged via TDM
// ---------------------------------------------------------------------------
__global__ __launch_bounds__(256) void gattn_mlp_gemm(
    const __bf16* __restrict__ A, const __bf16* __restrict__ Wb,
    const float* __restrict__ bias,
    __bf16* __restrict__ outb, float* __restrict__ outf, int relu) {
  int wid = blockIdx.x * 8 + (threadIdx.x >> 5);
  int lane = threadIdx.x & 31;
  int r0 = (wid >> 4) << 4;          // uniform across block
  int d0 = (wid & 15) << 4;

  __shared__ __bf16 as[16 * DD];     // 8 KB A band

#if defined(GATTN_HAS_TDM)
  if ((threadIdx.x >> 5) == 0) {
    // Tensor DMA: 2D tile, 16 rows x 128 dwords, contiguous into LDS
    unsigned lds_off = (unsigned)(uintptr_t)as;   // low 32 bits = LDS offset
    unsigned long long ga = (unsigned long long)(uintptr_t)(A + (size_t)r0 * DD);
    tdm_v4u g0;
    g0[0] = 1u;                                    // count=1, user mode
    g0[1] = lds_off;                               // lds_addr
    g0[2] = (unsigned)ga;                          // global_addr[31:0]
    g0[3] = (unsigned)((ga >> 32) & 0x1FFFFFFu) | (2u << 30);  // addr hi | type=2
    const unsigned dim0 = (DD * 2) / 4;            // 128 dwords per row
    const unsigned rows = 16;
    tdm_v8i g1;
    g1[0] = (int)(2u << 16);                       // data_size = 4B
    g1[1] = (int)((dim0 & 0xFFFFu) << 16);         // tensor_dim0 lo16 @ [63:48]
    g1[2] = (int)((dim0 >> 16) | (rows << 16));    // dim0 hi16 | tensor_dim1 lo16
    g1[3] = (int)(dim0 << 16);                     // dim1 hi16=0 | tile_dim0
    g1[4] = (int)rows;                             // tile_dim1 | tile_dim2=0
    g1[5] = (int)dim0;                             // tensor_dim0_stride[31:0]
    g1[6] = 0;                                     // stride0 hi | stride1 lo
    g1[7] = 0;
    tdm_v4i z4 = {0, 0, 0, 0};
#if defined(__clang_major__) && __clang_major__ >= 23
    tdm_v8i z8 = {0, 0, 0, 0, 0, 0, 0, 0};
    __builtin_amdgcn_tensor_load_to_lds(g0, g1, z4, z4, z8, 0);
#else
    __builtin_amdgcn_tensor_load_to_lds(g0, g1, z4, z4, 0);
#endif
    __builtin_amdgcn_s_wait_tensorcnt(0);
  }
  __syncthreads();
#else
  {
    const __bf16* gsrc = A + (size_t)r0 * DD;
    int i0 = threadIdx.x * 2;
#pragma unroll
    for (int t = 0; t < 2; ++t) {
      int idx = i0 + t;
      ((uint4*)as)[idx] = ((const uint4*)gsrc)[idx];
    }
  }
  __syncthreads();
#endif

  v8f acc = zero8();
#pragma unroll
  for (int ch = 0; ch < 8; ++ch) {
    v16bf af = load_a_frag(as, DD, 0, ch * 32, lane);
    v16bf bf = load_bT_frag(Wb, DD, ch * 32, d0, lane);
    acc = wmma_bf16(af, bf, acc);
  }
  int col = lane & 15, half = lane >> 4;
#pragma unroll
  for (int r = 0; r < 8; ++r) {
    int row = r0 + r + (half << 3);
    int cc = d0 + col;
    float v = acc[r] + bias[cc];
    if (relu) v = v > 0.f ? v : LEAK * v;
    if (outb) outb[(size_t)row * DD + cc] = (__bf16)v;
    if (outf) outf[(size_t)row * DD + cc] = v;
  }
}

// ---------------------------------------------------------------------------
// Host launcher
// ---------------------------------------------------------------------------
extern "C" void kernel_launch(void* const* d_in, const int* in_sizes, int n_in,
                              void* d_out, int out_size, void* d_ws, size_t ws_size,
                              hipStream_t stream) {
  const float* src   = (const float*)d_in[0];
  const float* x_lst = (const float*)d_in[1];
  const int*   adj   = (const int*)d_in[2];
  const float* dis   = (const float*)d_in[3];
  const float* Wq = (const float*)d_in[4];  const float* bq = (const float*)d_in[5];
  const float* Wk = (const float*)d_in[6];  const float* bk = (const float*)d_in[7];
  const float* Wv = (const float*)d_in[8];  const float* bv = (const float*)d_in[9];
  const float* W1 = (const float*)d_in[10]; const float* b1 = (const float*)d_in[11];
  const float* W2 = (const float*)d_in[12]; const float* b2 = (const float*)d_in[13];
  const float* W3 = (const float*)d_in[14]; const float* b3 = (const float*)d_in[15];
  const float* g1 = (const float*)d_in[16]; const float* be1 = (const float*)d_in[17];
  const float* g2 = (const float*)d_in[18]; const float* be2 = (const float*)d_in[19];

  float* h_out    = (float*)d_out;                        // [B,N,D]
  float* xlst_out = (float*)d_out + (size_t)BB * NN * DD; // [B,N,N]

  char* ws = (char*)d_ws;
  size_t off = 0;
  auto carve = [&](size_t bytes) {
    char* p = ws + off;
    off = (off + bytes + 255) & ~(size_t)255;
    return p;
  };
  __bf16* xpadT = (__bf16*)carve((size_t)BB * NPAD * DD * 2);
  __bf16* wpq   = (__bf16*)carve((size_t)DD * DKK * 2);
  __bf16* wpk   = (__bf16*)carve((size_t)DD * DKK * 2);
  __bf16* wpv   = (__bf16*)carve((size_t)DD * DKK * 2);
  __bf16* w1b   = (__bf16*)carve((size_t)DD * DD * 2);
  __bf16* w2b   = (__bf16*)carve((size_t)DD * DD * 2);
  __bf16* w3b   = (__bf16*)carve((size_t)DD * DD * 2);
  __bf16* qT    = (__bf16*)carve((size_t)BB * NN * DD * 2);
  __bf16* kT    = (__bf16*)carve((size_t)BB * NN * DD * 2);
  __bf16* vmat  = (__bf16*)carve((size_t)BB * DD * NN * 2);
  __bf16* attn  = (__bf16*)carve((size_t)BB * NN * NN * 2);
  float*  x1    = (float*)carve((size_t)BB * NN * DD * 4);  // reused as m3
  float*  hf    = (float*)carve((size_t)BB * NN * DD * 4);
  __bf16* hb    = (__bf16*)carve((size_t)BB * NN * DD * 2);
  __bf16* m1    = (__bf16*)carve((size_t)BB * NN * DD * 2);
  __bf16* m2    = (__bf16*)carve((size_t)BB * NN * DD * 2);
  float*  m3    = x1;  // x1 consumed by ln1 before m3 is produced

  // ---- pack ----
  gattn_pack_x<<<(BB * NPAD * DD) / 256, 256, 0, stream>>>(src, xpadT);
  gattn_pack_convw<<<(DD * DKK) / 256, 256, 0, stream>>>(Wq, wpq);
  gattn_pack_convw<<<(DD * DKK) / 256, 256, 0, stream>>>(Wk, wpk);
  gattn_pack_convw<<<(DD * DKK) / 256, 256, 0, stream>>>(Wv, wpv);
  gattn_pack_f2b<<<(DD * DD) / 256, 256, 0, stream>>>(W1, w1b);
  gattn_pack_f2b<<<(DD * DD) / 256, 256, 0, stream>>>(W2, w2b);
  gattn_pack_f2b<<<(DD * DD) / 256, 256, 0, stream>>>(W3, w3b);

  // ---- attention ----
  gattn_conv_qkv<<<1024, 256, 0, stream>>>(xpadT, wpq, wpk, wpv, bq, bk, bv,
                                           qT, kT, vmat);
  gattn_scores_softmax<<<BB * (NN / 16), 512, 0, stream>>>(qT, kT, dis, adj,
                                                           x_lst, attn, xlst_out);
  gattn_attn_v<<<1024, 256, 0, stream>>>(attn, vmat, x1);

  // ---- residual + norm1 ----
  gattn_layernorm<<<BB * NN, 256, 0, stream>>>(src, x1, g1, be1, hf, hb);

  // ---- MLP ----
  gattn_mlp_gemm<<<1024, 256, 0, stream>>>(hb, w1b, b1, m1, nullptr, 1);
  gattn_mlp_gemm<<<1024, 256, 0, stream>>>(m1, w2b, b2, m2, nullptr, 1);
  gattn_mlp_gemm<<<1024, 256, 0, stream>>>(m2, w3b, b3, nullptr, m3, 0);

  // ---- residual + norm2 -> output ----
  gattn_layernorm<<<BB * NN, 256, 0, stream>>>(hf, m3, g2, be2, h_out, nullptr);
}
